// EfficientAtt_9629316678153
// MI455X (gfx1250) — matrix operand order repaired and checked by reading, not compiled, not run on hardware
//
#include <hip/hip_runtime.h>
#include <hip/hip_bf16.h>
#include <stdint.h>

// ---------------------------------------------------------------------------
// Problem constants (from reference): B=8, C=256, H=W=64, N=4096,
// heads=8, Ch=24, sd=64, Ct=192.
// ---------------------------------------------------------------------------

typedef __attribute__((ext_vector_type(16))) _Float16 v16h;
typedef __attribute__((ext_vector_type(8)))  float    v8f;
typedef int v4i __attribute__((vector_size(16)));

union F16x16 { v16h h; uint32_t u[8]; };

__device__ __forceinline__ v8f zero8() {
    v8f z = {0.f, 0.f, 0.f, 0.f, 0.f, 0.f, 0.f, 0.f};
    return z;
}

__device__ __forceinline__ float hswish(float y) {
    float t = y + 3.0f;
    t = t < 0.0f ? 0.0f : (t > 6.0f ? 6.0f : t);
    return y * t * (1.0f / 6.0f);
}

// ---------------------------------------------------------------------------
// Cooperative global->LDS fill. Uses gfx1250 async-to-LDS path when the
// toolchain exposes the builtin (ASYNCcnt-tracked, per ISA 08_async_tensor §4),
// otherwise plain loads. Caller must __syncthreads() afterwards.
// The builtin's pointer params are v4i* in addrspace(1)/(3); we go through
// uintptr_t (low 32 bits of a flat LDS address are the LDS offset).
// ---------------------------------------------------------------------------
#if defined(__gfx1250__) && __has_builtin(__builtin_amdgcn_global_load_async_to_lds_b128)
#define HAVE_ASYNC_LDS 1
#endif

__device__ __forceinline__ void ldsfill(const float* __restrict__ src,
                                        float* dst_lds, int nfloats,
                                        int tid, int nthreads) {
#ifdef HAVE_ASYNC_LDS
    const int nvec = nfloats >> 2;                       // 16B per async op
    for (int i = tid; i < nvec; i += nthreads) {
        __builtin_amdgcn_global_load_async_to_lds_b128(
            (__attribute__((address_space(1))) v4i*)(uintptr_t)(src + i * 4),
            (__attribute__((address_space(3))) v4i*)(uintptr_t)(dst_lds + i * 4),
            0, 0);
    }
#if __has_builtin(__builtin_amdgcn_s_wait_asynccnt)
    __builtin_amdgcn_s_wait_asynccnt(0);
#else
    asm volatile("s_wait_asynccnt 0x0" ::: "memory");
#endif
#else
    for (int i = tid; i < nfloats; i += nthreads) dst_lds[i] = src[i];
#endif
}

// ---------------------------------------------------------------------------
// Core WMMA K-loop: one wave computes a 16(M) x 64(N) strip as 4 accumulators,
// reusing the A fragment across the 4 wmma ops of each K step.
//   A_mat[m][k] = A[m*256 + k]            (weights, row-major, ld=256)
//   B_mat[k][c] = Bm[c*256 + k]           (activations [n][k], ld=256)
// Fragment layouts: A 16x32, B 32x16 in contiguous k order (k = 16*(lane/16) + s); f32 C/D.
// ---------------------------------------------------------------------------
__device__ __forceinline__ void wmma_kloop4(const _Float16* __restrict__ A,
                                            const _Float16* __restrict__ Bm,
                                            int K, int lane, v8f acc[4]) {
    const int arow = lane & 15;
    const int akb  = (lane >> 4) * 16;   // A K sub-block (halves)
    const int bcol = lane & 15;
    const int bkb  = (lane >> 4) * 16;   // B K sub-block (halves)
    const uint32_t* Ap  = (const uint32_t*)(A + (long)arow * 256);
    const uint32_t* Bp0 = (const uint32_t*)(Bm + (long)bcol * 256);
    // consecutive B tiles: 16 columns apart = 16*256 halves = 2048 u32
    for (int k0 = 0; k0 < K; k0 += 32) {
        F16x16 af;
        const int ka = (k0 + akb) >> 1;
#pragma unroll
        for (int j = 0; j < 4; ++j) {
            af.u[j]     = Ap[ka + j];        // K = k0+akb+2j
            af.u[4 + j] = Ap[ka + 4 + j];    // K = k0+akb+8+2j
        }
        const int kb = (k0 + bkb) >> 1;
        F16x16 bf[4];
#pragma unroll
        for (int t = 0; t < 4; ++t) {
            const uint32_t* Bp = Bp0 + (long)t * 2048;
#pragma unroll
            for (int j = 0; j < 8; ++j) bf[t].u[j] = Bp[kb + j];
        }
        if (k0 + 32 < K) {                   // speculative prefetch, next K slab
            __builtin_prefetch((const void*)(Ap + ka + 16), 0, 3);
#pragma unroll
            for (int t = 0; t < 4; ++t)
                __builtin_prefetch((const void*)(Bp0 + (long)t * 2048 + kb + 16), 0, 3);
        }
#pragma unroll
        for (int t = 0; t < 4; ++t)
            acc[t] = __builtin_amdgcn_wmma_f32_16x16x32_f16(
                false, af.h, false, bf[t].h, (short)0, acc[t], false, false);
    }
}

// ---------------------------------------------------------------------------
// f32 -> f16 pack (grid-stride)
// ---------------------------------------------------------------------------
__global__ void pack_f16_kernel(const float* __restrict__ src,
                                _Float16* __restrict__ dst, long n) {
    long i = (long)blockIdx.x * blockDim.x + threadIdx.x;
    long stride = (long)gridDim.x * blockDim.x;
    for (; i < n; i += stride) dst[i] = (_Float16)src[i];
}

// ---------------------------------------------------------------------------
// Kernel 1: fused q/k/v input GEMM.
// xa[a][c][n] = sum_k W_slot[c][k] * in_a[n][k],  a in [0,24)
//   a<8  : slot0 = q_w,           input = seg
//   8-15 : slot1 = kv_w[0:256],   input = x   (K part)
//   16-23: slot2 = kv_w[256:512], input = x   (V part)
// One wave = 16(c) x 64(n); 8 waves/block. 24*16*64 = 24576 waves.
// ---------------------------------------------------------------------------
__global__ void qkv_gemm_kernel(const _Float16* __restrict__ xf16,
                                const _Float16* __restrict__ sf16,
                                const _Float16* __restrict__ wqkv,
                                float* __restrict__ xa) {
    const int wid  = blockIdx.x * 8 + (threadIdx.x >> 5);
    const int lane = threadIdx.x & 31;
    const int nt4   = wid & 63;
    const int ctile = (wid >> 6) & 15;
    const int a     = wid >> 10;                 // 0..23
    const int slot  = a >> 3;
    const _Float16* inp = (a < 8 ? sf16 : xf16) + (long)(a & 7) * 4096 * 256;
    const _Float16* A   = wqkv + ((long)slot * 256 + ctile * 16) * 256;
    const _Float16* Bm  = inp + (long)nt4 * 64 * 256;
    v8f acc[4];
#pragma unroll
    for (int t = 0; t < 4; ++t) acc[t] = zero8();
    wmma_kloop4(A, Bm, 256, lane, acc);
    const int col = lane & 15;
    const int mb  = (lane >> 4) * 8;
    float* outp = xa + ((long)a * 256 + ctile * 16) * 4096 + nt4 * 64 + col;
#pragma unroll
    for (int t = 0; t < 4; ++t)
#pragma unroll
        for (int r = 0; r < 8; ++r)
            outp[(long)(mb + r) * 4096 + t * 16] = acc[t][r];
}

// ---------------------------------------------------------------------------
// Generic depthwise conv, one thread per output element.
// mode 0: src = in + b*inBatchStride + c*4096   (channel-major, HW=64x64)
// mode 1: agg0 gather: src = xa[(c/64)*8 + b][192 + c%64]
// ---------------------------------------------------------------------------
__global__ void dwconv_kernel(const float* __restrict__ in, long inBatchStride,
                              const float* __restrict__ wgt,
                              const float* __restrict__ bias,
                              float* __restrict__ out, long outBatchStride,
                              int nBatch, int nCh, int ksz, int pad, int mode) {
    long tid = (long)blockIdx.x * blockDim.x + threadIdx.x;
    long total = (long)nBatch * nCh * 4096;
    if (tid >= total) return;
    const int n = (int)(tid & 4095);
    const int c = (int)((tid >> 12) % nCh);
    const int b = (int)(tid / ((long)nCh << 12));
    const float* src;
    if (mode == 0)
        src = in + b * inBatchStride + (long)c * 4096;
    else
        src = in + (((long)(c >> 6) * 8 + b) * 256 + 192 + (c & 63)) * 4096;
    const int h = n >> 6, w = n & 63;
    float acc = bias ? bias[c] : 0.0f;
    const float* wc = wgt + c * ksz * ksz;
    for (int ky = 0; ky < ksz; ++ky) {
        const int iy = h + ky - pad;
        if (iy < 0 || iy >= 64) continue;
        for (int kx = 0; kx < ksz; ++kx) {
            const int ix = w + kx - pad;
            if (ix < 0 || ix >= 64) continue;
            acc += src[iy * 64 + ix] * wc[ky * ksz + kx];
        }
    }
    out[b * outBatchStride + (long)c * 4096 + n] = acc;
}

// ---------------------------------------------------------------------------
// agg0 tail: 1x1 conv (192->64) + LayerNorm over 64 ch + hardswish,
// written directly into attnbuf[:, :, 192:256]. One thread per pixel.
// ---------------------------------------------------------------------------
__global__ void pw_agg0_ln_kernel(const float* __restrict__ dwt,   // (8,192,4096)
                                  const float* __restrict__ pw,    // (64,192)
                                  const float* __restrict__ g,
                                  const float* __restrict__ bta,
                                  float* __restrict__ attnbuf) {   // (8,4096,256)
    __shared__ float lw[64 * 192];
    ldsfill(pw, lw, 64 * 192, threadIdx.x, blockDim.x);
    __syncthreads();
    const int p = blockIdx.x * blockDim.x + threadIdx.x;   // [0, 32768)
    const int b = p >> 12, n = p & 4095;
    float acc[64];
#pragma unroll
    for (int co = 0; co < 64; ++co) acc[co] = 0.0f;
    const float* src = dwt + (long)b * 192 * 4096 + n;
    for (int ci = 0; ci < 192; ++ci) {
        const float v = src[(long)ci * 4096];
#pragma unroll
        for (int co = 0; co < 64; ++co) acc[co] += v * lw[co * 192 + ci];
    }
    float mu = 0.0f;
#pragma unroll
    for (int co = 0; co < 64; ++co) mu += acc[co];
    mu *= (1.0f / 64.0f);
    float var = 0.0f;
#pragma unroll
    for (int co = 0; co < 64; ++co) { float d = acc[co] - mu; var += d * d; }
    var *= (1.0f / 64.0f);
    const float inv = rsqrtf(var + 1e-5f);
    float* dst = attnbuf + (long)p * 256 + 192;
#pragma unroll
    for (int co = 0; co < 64; ++co)
        dst[co] = hswish((acc[co] - mu) * inv * g[co] + bta[co]);
}

// ---------------------------------------------------------------------------
// agg1/2/3 tail: 1x1 conv (64->64) + BatchNorm + hardswish -> xc channels.
// One thread per (a, pixel).
// ---------------------------------------------------------------------------
__global__ void pw_bn_kernel(const float* __restrict__ dwt,   // (24,64,4096)
                             const float* __restrict__ pw,    // (64,64)
                             const float* __restrict__ g,  const float* __restrict__ bb,
                             const float* __restrict__ mm, const float* __restrict__ vv,
                             float* __restrict__ xc, int chOffset) {
    __shared__ float lw[64 * 64];
    ldsfill(pw, lw, 64 * 64, threadIdx.x, blockDim.x);
    __syncthreads();
    const long p = (long)blockIdx.x * blockDim.x + threadIdx.x; // [0, 24*4096)
    const int a = (int)(p >> 12), n = (int)(p & 4095);
    float acc[64];
#pragma unroll
    for (int co = 0; co < 64; ++co) acc[co] = 0.0f;
    const float* src = dwt + (long)a * 64 * 4096 + n;
    for (int ci = 0; ci < 64; ++ci) {
        const float v = src[(long)ci * 4096];
#pragma unroll
        for (int co = 0; co < 64; ++co) acc[co] += v * lw[co * 64 + ci];
    }
    float* dst = xc + ((long)a * 192 + chOffset) * 4096 + n;
#pragma unroll
    for (int co = 0; co < 64; ++co) {
        const float scale = g[co] * rsqrtf(vv[co] + 1e-5f);
        dst[(long)co * 4096] = hswish((acc[co] - mm[co]) * scale + bb[co]);
    }
}

// ---------------------------------------------------------------------------
// Softmax stats over N for each k column (b, h*24+ch): block reduction.
// ---------------------------------------------------------------------------
__global__ void ksoftmax_stats_kernel(const float* __restrict__ xc,
                                      float* __restrict__ cmax,
                                      float* __restrict__ csum) {
    __shared__ float red[256];
    const int bc = blockIdx.x;               // b*192 + c, 1536 blocks
    const int b = bc / 192, c = bc % 192;
    const float* row = xc + ((long)(8 + b) * 192 + c) * 4096;
    float mx = -1e30f;
    for (int n = threadIdx.x; n < 4096; n += 256) mx = fmaxf(mx, row[n]);
    red[threadIdx.x] = mx; __syncthreads();
    for (int s = 128; s > 0; s >>= 1) {
        if (threadIdx.x < s) red[threadIdx.x] = fmaxf(red[threadIdx.x], red[threadIdx.x + s]);
        __syncthreads();
    }
    mx = red[0]; __syncthreads();
    float sm = 0.0f;
    for (int n = threadIdx.x; n < 4096; n += 256) sm += __expf(row[n] - mx);
    red[threadIdx.x] = sm; __syncthreads();
    for (int s = 128; s > 0; s >>= 1) {
        if (threadIdx.x < s) red[threadIdx.x] += red[threadIdx.x + s];
        __syncthreads();
    }
    if (threadIdx.x == 0) { cmax[bc] = mx; csum[bc] = red[0]; }
}

// ---------------------------------------------------------------------------
// ktv[b][h][kc][vc] = sum_n softmax(k)[n,kc] * v[n,vc].
// Block per (b, h, 4-kc group): 8*8*6 = 384 blocks. Each thread carries a
// 4x24 accumulator; wave32 shuffle reduction then cross-wave LDS combine.
// V rows are re-read only 6x (vs 24x for block-per-kc).
// ---------------------------------------------------------------------------
__global__ void ktv_kernel(const float* __restrict__ xc,
                           const float* __restrict__ cmax,
                           const float* __restrict__ csum,
                           float* __restrict__ ktv) {
    __shared__ float part[8 * 96];
    const int idx = blockIdx.x;              // (b*8 + h)*6 + kcg
    const int kcg = idx % 6;
    const int h   = (idx / 6) % 8;
    const int b   = idx / 48;
    const float* kbase = xc + ((long)(8 + b) * 192 + h * 24 + kcg * 4) * 4096;
    const float* vbase = xc + ((long)(16 + b) * 192 + h * 24) * 4096;
    float mx[4], isum[4];
#pragma unroll
    for (int kcl = 0; kcl < 4; ++kcl) {
        const int cc = b * 192 + h * 24 + kcg * 4 + kcl;
        mx[kcl]   = cmax[cc];
        isum[kcl] = 1.0f / csum[cc];
    }
    float acc[96];
#pragma unroll
    for (int i = 0; i < 96; ++i) acc[i] = 0.0f;
    for (int n = threadIdx.x; n < 4096; n += 256) {
        float e[4];
#pragma unroll
        for (int kcl = 0; kcl < 4; ++kcl)
            e[kcl] = __expf(kbase[(long)kcl * 4096 + n] - mx[kcl]);
#pragma unroll
        for (int vc = 0; vc < 24; ++vc) {
            const float v = vbase[(long)vc * 4096 + n];
#pragma unroll
            for (int kcl = 0; kcl < 4; ++kcl)
                acc[kcl * 24 + vc] += e[kcl] * v;
        }
    }
    const int lane = threadIdx.x & 31;
    const int wv   = threadIdx.x >> 5;
#pragma unroll
    for (int i = 0; i < 96; ++i) {
        float v = acc[i];
        for (int off = 16; off > 0; off >>= 1) v += __shfl_down(v, off, 32);
        if (lane == 0) part[wv * 96 + i] = v;
    }
    __syncthreads();
    if (threadIdx.x < 96) {
        float s = 0.0f;
#pragma unroll
        for (int w = 0; w < 8; ++w) s += part[w * 96 + threadIdx.x];
        const int kcl = threadIdx.x / 24, vc = threadIdx.x % 24;
        const int kc  = kcg * 4 + kcl;
        ktv[((long)(b * 8 + h) * 24 + kc) * 24 + vc] = s * isum[kcl];
    }
}

// ---------------------------------------------------------------------------
// eff_att + crpe combine: attnbuf[b][n][h*24+ch] =
//    scale * sum_kc q[kc]*ktv[kc][ch]  +  q[ch] * conv_v[ch].
// Block per (b, h, 256-pixel chunk); ktv tile cached in LDS.
// ---------------------------------------------------------------------------
__global__ void attn_assemble_kernel(const float* __restrict__ xc,
                                     const float* __restrict__ ktv,
                                     const float* __restrict__ convv,
                                     float* __restrict__ attnbuf) {
    __shared__ float kt[24 * 24];
    const int nchunk = blockIdx.x & 15;
    const int h = (blockIdx.x >> 4) & 7;
    const int b = blockIdx.x >> 7;
    const float* kt_src = ktv + (long)(b * 8 + h) * 24 * 24;
    for (int i = threadIdx.x; i < 576; i += blockDim.x) kt[i] = kt_src[i];
    __syncthreads();
    const int n = nchunk * 256 + threadIdx.x;
    const float* qbase = xc    + ((long)b * 192 + h * 24) * 4096 + n;
    const float* cvb   = convv + ((long)b * 192 + h * 24) * 4096 + n;
    float qv[24];
#pragma unroll
    for (int kc = 0; kc < 24; ++kc) qv[kc] = qbase[(long)kc * 4096];
    const float scale = 0.17677669529663687f;   // (C/heads)^-0.5 = 1/sqrt(32)
    float* dst = attnbuf + ((long)b * 4096 + n) * 256 + h * 24;
#pragma unroll
    for (int ch = 0; ch < 24; ++ch) {
        float s = 0.0f;
#pragma unroll
        for (int kc = 0; kc < 24; ++kc) s += qv[kc] * kt[kc * 24 + ch];
        dst[ch] = scale * s + qv[ch] * cvb[(long)ch * 4096];
    }
}

// ---------------------------------------------------------------------------
// Final projection GEMM: out[b][n][co] = attn[b][n][:] . proj_w[co][:] + b[co]
// One wave = 16(co) x 64(n); 8*16*64 = 8192 waves.
// ---------------------------------------------------------------------------
__global__ void proj_gemm_kernel(const _Float16* __restrict__ attn16,
                                 const _Float16* __restrict__ pw16,
                                 const float* __restrict__ pb,
                                 float* __restrict__ out) {
    const int wid  = blockIdx.x * 8 + (threadIdx.x >> 5);
    const int lane = threadIdx.x & 31;
    const int nt4   = wid & 63;
    const int ctile = (wid >> 6) & 15;
    const int b     = wid >> 10;             // 0..7
    const _Float16* A  = pw16 + (long)ctile * 16 * 256;
    const _Float16* Bm = attn16 + ((long)b * 4096 + nt4 * 64) * 256;
    v8f acc[4];
#pragma unroll
    for (int t = 0; t < 4; ++t) acc[t] = zero8();
    wmma_kloop4(A, Bm, 256, lane, acc);
    const int col = lane & 15;
    const int mb  = (lane >> 4) * 8;
#pragma unroll
    for (int t = 0; t < 4; ++t) {
        float* outp = out + ((long)b * 4096 + nt4 * 64 + t * 16 + col) * 256 + ctile * 16;
#pragma unroll
        for (int r = 0; r < 8; ++r)
            outp[mb + r] = acc[t][r] + pb[ctile * 16 + mb + r];
    }
}

// ---------------------------------------------------------------------------
// Host-side launcher
// ---------------------------------------------------------------------------
extern "C" void kernel_launch(void* const* d_in, const int* in_sizes, int n_in,
                              void* d_out, int out_size, void* d_ws, size_t ws_size,
                              hipStream_t stream) {
    (void)in_sizes; (void)n_in; (void)out_size; (void)ws_size;
    const float* x       = (const float*)d_in[0];
    const float* seg     = (const float*)d_in[1];
    const float* kv_w    = (const float*)d_in[2];
    const float* q_w     = (const float*)d_in[3];
    const float* proj_w  = (const float*)d_in[4];
    const float* proj_b  = (const float*)d_in[5];
    const float* agg0_dw = (const float*)d_in[6];
    const float* agg0_pw = (const float*)d_in[7];
    const float* ln_g    = (const float*)d_in[8];
    const float* ln_b    = (const float*)d_in[9];
    const float* agg1_dw = (const float*)d_in[10];
    const float* agg1_pw = (const float*)d_in[11];
    const float* bn1_g   = (const float*)d_in[12];
    const float* bn1_b   = (const float*)d_in[13];
    const float* bn1_m   = (const float*)d_in[14];
    const float* bn1_v   = (const float*)d_in[15];
    const float* agg2_dw = (const float*)d_in[16];
    const float* agg2_pw = (const float*)d_in[17];
    const float* bn2_g   = (const float*)d_in[18];
    const float* bn2_b   = (const float*)d_in[19];
    const float* bn2_m   = (const float*)d_in[20];
    const float* bn2_v   = (const float*)d_in[21];
    const float* agg3_dw = (const float*)d_in[22];
    const float* agg3_pw = (const float*)d_in[23];
    const float* bn3_g   = (const float*)d_in[24];
    const float* bn3_b   = (const float*)d_in[25];
    const float* bn3_m   = (const float*)d_in[26];
    const float* bn3_v   = (const float*)d_in[27];
    const float* crpe_w3 = (const float*)d_in[28];
    const float* crpe_b3 = (const float*)d_in[29];
    const float* crpe_w5 = (const float*)d_in[30];
    const float* crpe_b5 = (const float*)d_in[31];
    const float* crpe_w7 = (const float*)d_in[32];
    const float* crpe_b7 = (const float*)d_in[33];

    // workspace carve-up (256B aligned)
    char* ws = (char*)d_ws;
    auto alloc = [&](size_t bytes) -> char* {
        char* p = ws; ws += (bytes + 255) & ~(size_t)255; return p;
    };
    float*    xa      = (float*)alloc(24L * 256 * 4096 * 4);   // qkv, channel-major
    float*    xc      = (float*)alloc(24L * 192 * 4096 * 4);   // post-agg q/k/v
    float*    dwtmp   = (float*)alloc(8L * 192 * 4096 * 4);    // dw-conv scratch
    float*    convv   = (float*)alloc(8L * 192 * 4096 * 4);    // crpe conv output
    float*    attnbuf = (float*)alloc(8L * 4096 * 256 * 4);    // pre-projection
    _Float16* xf16    = (_Float16*)alloc(8L * 4096 * 256 * 2);
    _Float16* sf16    = (_Float16*)alloc(8L * 4096 * 256 * 2);
    _Float16* attn16  = (_Float16*)alloc(8L * 4096 * 256 * 2);
    _Float16* wqkv16  = (_Float16*)alloc(3L * 256 * 256 * 2);  // [q; kv_k; kv_v]
    _Float16* pjw16   = (_Float16*)alloc(256L * 256 * 2);
    float*    cmax    = (float*)alloc(1536 * 4);
    float*    csum    = (float*)alloc(1536 * 4);
    float*    ktvb    = (float*)alloc(8L * 8 * 24 * 24 * 4);

    const dim3 blk(256);
    const long nelem = 8L * 4096 * 256;

    // 1) f16 staging of GEMM operands
    pack_f16_kernel<<<2048, blk, 0, stream>>>(x,      xf16, nelem);
    pack_f16_kernel<<<2048, blk, 0, stream>>>(seg,    sf16, nelem);
    pack_f16_kernel<<<64,   blk, 0, stream>>>(q_w,    wqkv16,              256L * 256);
    pack_f16_kernel<<<128,  blk, 0, stream>>>(kv_w,   wqkv16 + 256L * 256, 512L * 256);
    pack_f16_kernel<<<64,   blk, 0, stream>>>(proj_w, pjw16,               256L * 256);

    // 2) q/k/v input GEMM (WMMA): 24576 waves, 8 waves/block
    qkv_gemm_kernel<<<3072, blk, 0, stream>>>(xf16, sf16, wqkv16, xa);

    // 3) agg0 path -> attnbuf[:, :, 192:256]
    dwconv_kernel<<<24576, blk, 0, stream>>>(xa, 0, agg0_dw, nullptr,
                                             dwtmp, 192L * 4096, 8, 192, 3, 1, 1);
    pw_agg0_ln_kernel<<<128, blk, 0, stream>>>(dwtmp, agg0_pw, ln_g, ln_b, attnbuf);

    // 4) agg1/2/3 -> xc channels [0:64), [64:128), [128:192)
    dwconv_kernel<<<24576, blk, 0, stream>>>(xa, 256L * 4096, agg1_dw, nullptr,
                                             dwtmp, 64L * 4096, 24, 64, 3, 1, 0);
    pw_bn_kernel<<<384, blk, 0, stream>>>(dwtmp, agg1_pw, bn1_g, bn1_b, bn1_m, bn1_v, xc, 0);
    dwconv_kernel<<<24576, blk, 0, stream>>>(xa + 64L * 4096, 256L * 4096, agg2_dw, nullptr,
                                             dwtmp, 64L * 4096, 24, 64, 5, 2, 0);
    pw_bn_kernel<<<384, blk, 0, stream>>>(dwtmp, agg2_pw, bn2_g, bn2_b, bn2_m, bn2_v, xc, 64);
    dwconv_kernel<<<24576, blk, 0, stream>>>(xa + 128L * 4096, 256L * 4096, agg3_dw, nullptr,
                                             dwtmp, 64L * 4096, 24, 64, 7, 3, 0);
    pw_bn_kernel<<<384, blk, 0, stream>>>(dwtmp, agg3_pw, bn3_g, bn3_b, bn3_m, bn3_v, xc, 128);

    // 5) linear attention: softmax stats over N, then K^T V (24x24 per head)
    ksoftmax_stats_kernel<<<1536, blk, 0, stream>>>(xc, cmax, csum);
    ktv_kernel<<<384, blk, 0, stream>>>(xc, cmax, csum, ktvb);

    // 6) crpe depthwise convs on the V image (xc batches 16..23)
    const float* vimg = xc + 16L * 192 * 4096;
    dwconv_kernel<<<6144, blk, 0, stream>>>(vimg,              192L * 4096, crpe_w3, crpe_b3,
                                            convv,             192L * 4096, 8, 48, 3, 1, 0);
    dwconv_kernel<<<9216, blk, 0, stream>>>(vimg + 48L * 4096, 192L * 4096, crpe_w5, crpe_b5,
                                            convv + 48L * 4096, 192L * 4096, 8, 72, 5, 2, 0);
    dwconv_kernel<<<9216, blk, 0, stream>>>(vimg + 120L * 4096, 192L * 4096, crpe_w7, crpe_b7,
                                            convv + 120L * 4096, 192L * 4096, 8, 72, 7, 3, 0);

    // 7) combine eff_att + crpe -> attnbuf[:, :, 0:192]
    attn_assemble_kernel<<<1024, blk, 0, stream>>>(xc, ktvb, convv, attnbuf);

    // 8) final projection (WMMA)
    pack_f16_kernel<<<2048, blk, 0, stream>>>(attnbuf, attn16, nelem);
    proj_gemm_kernel<<<1024, blk, 0, stream>>>(attn16, pjw16, proj_b, (float*)d_out);
}
